// PLM_GATNet_24507083391734
// MI455X (gfx1250) — matrix operand order repaired and checked by reading, not compiled, not run on hardware
//
#include <hip/hip_runtime.h>
#include <hip/hip_bf16.h>

typedef __attribute__((ext_vector_type(16))) __bf16 v16bf;
typedef __attribute__((ext_vector_type(8)))  __bf16 v8bf;
typedef __attribute__((ext_vector_type(8)))  float  v8f;

#define N_NODES 50000
#define N_EDGES 400000
#define N_ETOT  450000      /* edges + self loops */
#define NB      256
#define F_IN    78
#define NH1     10
#define F1      78
#define HF1     780         /* NH1*F1 */
#define KP1     96          /* F_IN padded to 3*32 */
#define LD1     784         /* HF1 padded to 49*16 */
#define KP2     800         /* HF1 padded to 25*32 */
#define F2      128
#define EMB     320
#define PLMD    128
#define BN_EPS  1e-5f

// ---------------------------------------------------------------- helpers
static __device__ __forceinline__ void atomic_max_f32(float* addr, float v) {
  // sign-aware float max via int atomics (valid IEEE ordering trick)
  if (v >= 0.f) atomicMax((int*)addr, __float_as_int(v));
  else          atomicMin((unsigned int*)addr, __float_as_uint(v));
}

static __device__ __forceinline__ void atomic_pk_add_bf16(__bf16* addr, float lo, float hi) {
  unsigned short l = __builtin_bit_cast(unsigned short, (__bf16)lo);
  unsigned short h = __builtin_bit_cast(unsigned short, (__bf16)hi);
  unsigned int data = (unsigned int)l | ((unsigned int)h << 16);
  asm volatile("global_atomic_pk_add_bf16 %0, %1, off"
               :: "v"(addr), "v"(data) : "memory");
}

// ---------------------------------------------------------------- packing
__global__ void pack_x_k(const float* __restrict__ x, __bf16* __restrict__ XB) {
  size_t idx = (size_t)blockIdx.x * blockDim.x + threadIdx.x;
  if (idx >= (size_t)N_NODES * KP1) return;
  int n = (int)(idx / KP1), k = (int)(idx % KP1);
  XB[idx] = (k < F_IN) ? (__bf16)x[(size_t)n * F_IN + k] : (__bf16)0.f;
}

__global__ void pack_w1_k(const float* __restrict__ W1, __bf16* __restrict__ WT) {
  size_t idx = (size_t)blockIdx.x * blockDim.x + threadIdx.x;
  if (idx >= (size_t)LD1 * KP1) return;
  int c = (int)(idx / KP1), k = (int)(idx % KP1);
  WT[idx] = (c < HF1 && k < F_IN) ? (__bf16)W1[(size_t)k * HF1 + c] : (__bf16)0.f;
}

__global__ void pack_w2_k(const float* __restrict__ W2, __bf16* __restrict__ WT) {
  size_t idx = (size_t)blockIdx.x * blockDim.x + threadIdx.x;
  if (idx >= (size_t)F2 * KP2) return;
  int c = (int)(idx / KP2), k = (int)(idx % KP2);
  WT[idx] = (k < HF1) ? (__bf16)W2[(size_t)k * F2 + c] : (__bf16)0.f;
}

__global__ void fill_f32_k(float* __restrict__ p, float v, size_t n) {
  size_t i = (size_t)blockIdx.x * blockDim.x + threadIdx.x;
  if (i < n) p[i] = v;
}

// ---------------------------------------------------------------- WMMA GEMM
// C[M x ncols] = A[M x K] * B[K x ncols]; A row-major bf16 (lda), B supplied
// pre-transposed as Bt[ncols x K] (ldb); output stored bf16 (ldc).
// Block = 8 waves, one 16-col N tile: the 16 x K weight slab is staged ONCE
// into LDS with async global->LDS copies (ASYNCcnt) and reused by all 8 waves
// (8 M tiles), cutting B global traffic 8x. A streams from global (clauses +
// prefetch). K consumed 32/step with v_wmma_f32_16x16x32_bf16 (f32 accum).
__global__ void __launch_bounds__(256) wmma_gemm_bf16(
    const __bf16* __restrict__ A, const __bf16* __restrict__ Bt,
    __bf16* __restrict__ C, int mtiles, int ntiles, int ktiles,
    int lda, int ldb, int ldc)
{
  extern __shared__ __bf16 sB[];            // [16 cols][ldb] weight slab
  int nt = blockIdx.x % ntiles;
  int mb = blockIdx.x / ntiles;

  // ---- async stage B tile (16 columns x full padded K, contiguous in Bt)
  {
    int chunks = (16 * ldb * 2) / 16;       // 16-byte chunks
    const char* gsrc = (const char*)(Bt + (size_t)nt * 16 * ldb);
    char* lbase = (char*)sB;
    for (int i = threadIdx.x; i < chunks; i += 256) {
      unsigned loff = (unsigned)(uintptr_t)(lbase + i * 16);
      const char* g = gsrc + (size_t)i * 16;
      asm volatile("global_load_async_to_lds_b128 %0, %1, off"
                   :: "v"(loff), "v"(g) : "memory");
    }
    asm volatile("s_wait_asynccnt 0x0" ::: "memory");
  }
  __syncthreads();

  int wave = threadIdx.x >> 5;
  int lane = threadIdx.x & 31;
  int mt = mb * 8 + wave;                   // wave-uniform guard: EXEC stays all-1
  if (mt < mtiles) {
    int half = lane >> 4;                   // K-half select per ISA operand layout
    int l15  = lane & 15;
    const __bf16* arow = A  + (size_t)(mt * 16 + l15) * lda + half * 8;
    const __bf16* bcol = sB + (size_t)l15 * ldb + half * 8;
    v8f c = {};
    for (int kt = 0; kt < ktiles; ++kt) {
      int k0 = kt * 32;
      v8bf a0 = *(const v8bf*)(arow + k0);       // K = k0 + half*8 + [0..7]
      v8bf a1 = *(const v8bf*)(arow + k0 + 16);  // K = k0+16+half*8 + [0..7]
      v8bf b0 = *(const v8bf*)(bcol + k0);       // K = k0 + half*8 + [0..7]
      v8bf b1 = *(const v8bf*)(bcol + k0 + 16);  // K = k0+16+half*8 + [0..7]
      __builtin_prefetch(arow + k0 + 32, 0, 3);  // global_prefetch_b8
      v16bf a = __builtin_shufflevector(a0, a1, 0,1,2,3,4,5,6,7,8,9,10,11,12,13,14,15);
      v16bf b = __builtin_shufflevector(b0, b1, 0,1,2,3,4,5,6,7,8,9,10,11,12,13,14,15);
      c = __builtin_amdgcn_wmma_f32_16x16x32_bf16(false, a, false, b, (short)0, c, false, false);
    }
    int col   = nt * 16 + l15;
    int rbase = mt * 16 + half * 8;              // C layout: vgpr r -> M=r (+8 hi lanes)
    #pragma unroll
    for (int r = 0; r < 8; ++r)
      C[(size_t)(rbase + r) * ldc + col] = (__bf16)c[r];
  }
}

// ---------------------------------------------------------------- GAT conv 1
__global__ void dots1_k(const __bf16* __restrict__ h1b, const float* __restrict__ a1s,
                        const float* __restrict__ a1d, float* __restrict__ ssrc,
                        float* __restrict__ sdst) {
  size_t idx = (size_t)blockIdx.x * blockDim.x + threadIdx.x;
  if (idx >= (size_t)N_NODES * NH1) return;
  int n = (int)(idx / NH1), h = (int)(idx % NH1);
  const __bf16* hp = h1b + (size_t)n * LD1 + h * F1;
  const float* as = a1s + h * F1;
  const float* ad = a1d + h * F1;
  float ss = 0.f, sd = 0.f;
  for (int f = 0; f < F1; ++f) { float v = (float)hp[f]; ss += v * as[f]; sd += v * ad[f]; }
  ssrc[idx] = ss; sdst[idx] = sd;
}

__global__ void scores1_k(const int* __restrict__ ei, const float* __restrict__ ssrc,
                          const float* __restrict__ sdst, float* __restrict__ esc,
                          float* __restrict__ m1) {
  size_t idx = (size_t)blockIdx.x * blockDim.x + threadIdx.x;
  if (idx >= (size_t)N_ETOT * NH1) return;
  int e = (int)(idx / NH1), h = (int)(idx % NH1);
  int src = (e < N_EDGES) ? ei[e]           : (e - N_EDGES);
  int dst = (e < N_EDGES) ? ei[N_EDGES + e] : (e - N_EDGES);
  float s = ssrc[(size_t)src * NH1 + h] + sdst[(size_t)dst * NH1 + h];
  s = (s > 0.f) ? s : 0.2f * s;                       // leaky_relu 0.2
  esc[idx] = s;
  atomic_max_f32(&m1[(size_t)dst * NH1 + h], s);
}

__global__ void exp1_k(const int* __restrict__ ei, float* __restrict__ esc,
                       const float* __restrict__ m1, float* __restrict__ den) {
  size_t idx = (size_t)blockIdx.x * blockDim.x + threadIdx.x;
  if (idx >= (size_t)N_ETOT * NH1) return;
  int e = (int)(idx / NH1), h = (int)(idx % NH1);
  int dst = (e < N_EDGES) ? ei[N_EDGES + e] : (e - N_EDGES);
  float p = __expf(esc[idx] - m1[(size_t)dst * NH1 + h]);
  esc[idx] = p;
  atomicAdd(&den[(size_t)dst * NH1 + h], p);
}

__global__ void agg1_k(const int* __restrict__ ei, const float* __restrict__ esc,
                       const float* __restrict__ den, const __bf16* __restrict__ h1b,
                       __bf16* __restrict__ agg) {
  size_t idx = (size_t)blockIdx.x * blockDim.x + threadIdx.x;
  if (idx >= (size_t)N_ETOT * (HF1 / 2)) return;
  int e = (int)(idx / (HF1 / 2));
  int j = (int)(idx % (HF1 / 2)) * 2;        // pair; never straddles a head (78 even)
  int h = j / F1;
  int src = (e < N_EDGES) ? ei[e]           : (e - N_EDGES);
  int dst = (e < N_EDGES) ? ei[N_EDGES + e] : (e - N_EDGES);
  float alpha = esc[(size_t)e * NH1 + h] / den[(size_t)dst * NH1 + h];
  float v0 = alpha * (float)h1b[(size_t)src * LD1 + j];
  float v1 = alpha * (float)h1b[(size_t)src * LD1 + j + 1];
  atomic_pk_add_bf16(agg + (size_t)dst * LD1 + j, v0, v1);
}

__global__ void finalize1_k(const __bf16* __restrict__ agg, const float* __restrict__ b1,
                            __bf16* __restrict__ out1b) {
  size_t idx = (size_t)blockIdx.x * blockDim.x + threadIdx.x;
  if (idx >= (size_t)N_NODES * KP2) return;
  int n = (int)(idx / KP2), j = (int)(idx % KP2);
  float v = 0.f;
  if (j < HF1) {
    float a = (float)agg[(size_t)n * LD1 + j] + b1[j];
    v = (a > 0.f) ? a : (__expf(a) - 1.f);            // elu
  }
  out1b[idx] = (__bf16)v;
}

// ---------------------------------------------------------------- GAT conv 2
__global__ void dots2_k(const __bf16* __restrict__ h2b, const float* __restrict__ a2s,
                        const float* __restrict__ a2d, float* __restrict__ s2s,
                        float* __restrict__ s2d) {
  int n = blockIdx.x * blockDim.x + threadIdx.x;
  if (n >= N_NODES) return;
  const __bf16* hp = h2b + (size_t)n * F2;
  float ss = 0.f, sd = 0.f;
  for (int f = 0; f < F2; ++f) { float v = (float)hp[f]; ss += v * a2s[f]; sd += v * a2d[f]; }
  s2s[n] = ss; s2d[n] = sd;
}

__global__ void scores2_k(const int* __restrict__ ei, const float* __restrict__ s2s,
                          const float* __restrict__ s2d, float* __restrict__ esc,
                          float* __restrict__ m2) {
  int e = blockIdx.x * blockDim.x + threadIdx.x;
  if (e >= N_ETOT) return;
  int src = (e < N_EDGES) ? ei[e]           : (e - N_EDGES);
  int dst = (e < N_EDGES) ? ei[N_EDGES + e] : (e - N_EDGES);
  float s = s2s[src] + s2d[dst];
  s = (s > 0.f) ? s : 0.2f * s;
  esc[e] = s;
  atomic_max_f32(&m2[dst], s);
}

__global__ void exp2_k(const int* __restrict__ ei, float* __restrict__ esc,
                       const float* __restrict__ m2, float* __restrict__ den) {
  int e = blockIdx.x * blockDim.x + threadIdx.x;
  if (e >= N_ETOT) return;
  int dst = (e < N_EDGES) ? ei[N_EDGES + e] : (e - N_EDGES);
  float p = __expf(esc[e] - m2[dst]);
  esc[e] = p;
  atomicAdd(&den[dst], p);
}

__global__ void agg2_k(const int* __restrict__ ei, const float* __restrict__ esc,
                       const float* __restrict__ den, const __bf16* __restrict__ h2b,
                       float* __restrict__ agg) {
  size_t idx = (size_t)blockIdx.x * blockDim.x + threadIdx.x;
  if (idx >= (size_t)N_ETOT * F2) return;
  int e = (int)(idx / F2), f = (int)(idx % F2);
  int src = (e < N_EDGES) ? ei[e]           : (e - N_EDGES);
  int dst = (e < N_EDGES) ? ei[N_EDGES + e] : (e - N_EDGES);
  float alpha = esc[e] / den[dst];
  atomicAdd(&agg[(size_t)dst * F2 + f], alpha * (float)h2b[(size_t)src * F2 + f]);
}

// relu(+bias) then global-max pool by graph id (values >= 0 -> int-max trick, g init 0)
__global__ void pool_k(const float* __restrict__ agg, const float* __restrict__ b2,
                       const int* __restrict__ batch, float* __restrict__ g) {
  size_t idx = (size_t)blockIdx.x * blockDim.x + threadIdx.x;
  if (idx >= (size_t)N_NODES * F2) return;
  int n = (int)(idx / F2), f = (int)(idx % F2);
  float v = fmaxf(agg[idx] + b2[f], 0.f);
  atomicMax((int*)&g[(size_t)batch[n] * F2 + f], __float_as_int(v));
}

// ---------------------------------------------------------------- head MLP
__global__ void fc_relu_k(const float* __restrict__ in, const float* __restrict__ W,
                          const float* __restrict__ b, float* __restrict__ outp,
                          int rows, int K, int Ncols) {
  int idx = blockIdx.x * blockDim.x + threadIdx.x;
  if (idx >= rows * Ncols) return;
  int r = idx / Ncols, c = idx % Ncols;
  float acc = b[c];
  const float* ip = in + (size_t)r * K;
  for (int k = 0; k < K; ++k) acc += ip[k] * W[(size_t)k * Ncols + c];
  outp[idx] = fmaxf(acc, 0.f);
}

__global__ void plm_k(const float* __restrict__ te, const float* __restrict__ Wxt,
                      const float* __restrict__ bxt, const float* __restrict__ gamma,
                      const float* __restrict__ beta, const float* __restrict__ rmean,
                      const float* __restrict__ rvar, float* __restrict__ xt) {
  int idx = blockIdx.x * blockDim.x + threadIdx.x;
  if (idx >= NB * PLMD) return;
  int r = idx / PLMD, c = idx % PLMD;
  float acc = bxt[c];
  for (int k = 0; k < EMB; ++k) acc += te[(size_t)r * EMB + k] * Wxt[(size_t)k * PLMD + c];
  float v = (acc - rmean[c]) * rsqrtf(rvar[c] + BN_EPS) * gamma[c] + beta[c];
  xt[idx] = fmaxf(v, 0.f);
}

__global__ void concat_k(const float* __restrict__ g, const float* __restrict__ xt,
                         float* __restrict__ xcat) {
  int idx = blockIdx.x * blockDim.x + threadIdx.x;
  if (idx >= NB * 256) return;
  int r = idx / 256, k = idx % 256;
  xcat[idx] = (k < F2) ? g[(size_t)r * F2 + k] : xt[(size_t)r * PLMD + (k - F2)];
}

__global__ void out_k(const float* __restrict__ xc2, const float* __restrict__ Wo,
                      const float* __restrict__ bo, float* __restrict__ outp) {
  int r = blockIdx.x * blockDim.x + threadIdx.x;
  if (r >= NB) return;
  float acc = bo[0];
  for (int k = 0; k < 256; ++k) acc += xc2[(size_t)r * 256 + k] * Wo[k];
  outp[r] = acc;
}

// ---------------------------------------------------------------- launch
static inline unsigned gridFor(size_t n, unsigned b) { return (unsigned)((n + b - 1) / b); }

extern "C" void kernel_launch(void* const* d_in, const int* in_sizes, int n_in,
                              void* d_out, int out_size, void* d_ws, size_t ws_size,
                              hipStream_t stream) {
  (void)in_sizes; (void)n_in; (void)out_size; (void)ws_size;
  const float* x     = (const float*)d_in[0];
  const int*   ei    = (const int*)d_in[1];
  const int*   batch = (const int*)d_in[2];
  const float* te    = (const float*)d_in[3];
  const float* W1    = (const float*)d_in[4];
  const float* a1s   = (const float*)d_in[5];
  const float* a1d   = (const float*)d_in[6];
  const float* b1    = (const float*)d_in[7];
  const float* W2    = (const float*)d_in[8];
  const float* a2s   = (const float*)d_in[9];
  const float* a2d   = (const float*)d_in[10];
  const float* b2    = (const float*)d_in[11];
  const float* Wg    = (const float*)d_in[12];
  const float* bg    = (const float*)d_in[13];
  const float* Wxt   = (const float*)d_in[14];
  const float* bxt   = (const float*)d_in[15];
  const float* gamma = (const float*)d_in[16];
  const float* beta  = (const float*)d_in[17];
  const float* rmean = (const float*)d_in[18];
  const float* rvar  = (const float*)d_in[19];
  const float* Wf1   = (const float*)d_in[20];
  const float* bf1   = (const float*)d_in[21];
  const float* Wf2   = (const float*)d_in[22];
  const float* bf2   = (const float*)d_in[23];
  const float* Wo    = (const float*)d_in[24];
  const float* bo    = (const float*)d_in[25];
  float* out = (float*)d_out;

  char* ws = (char*)d_ws;
  size_t off = 0;
  auto alloc = [&](size_t bytes) { size_t o = off; off += (bytes + 255) & ~(size_t)255; return o; };

  __bf16* XB    = (__bf16*)(ws + alloc((size_t)N_NODES * KP1 * 2));
  __bf16* WT1   = (__bf16*)(ws + alloc((size_t)LD1 * KP1 * 2));
  __bf16* h1b   = (__bf16*)(ws + alloc((size_t)N_NODES * LD1 * 2));
  float*  ssrc1 = (float*) (ws + alloc((size_t)N_NODES * NH1 * 4));
  float*  sdst1 = (float*) (ws + alloc((size_t)N_NODES * NH1 * 4));
  float*  m1    = (float*) (ws + alloc((size_t)N_NODES * NH1 * 4));
  float*  den1  = (float*) (ws + alloc((size_t)N_NODES * NH1 * 4));
  float*  esc1  = (float*) (ws + alloc((size_t)N_ETOT * NH1 * 4));
  __bf16* agg1  = (__bf16*)(ws + alloc((size_t)N_NODES * LD1 * 2));
  __bf16* out1b = (__bf16*)(ws + alloc((size_t)N_NODES * KP2 * 2));
  __bf16* WT2   = (__bf16*)(ws + alloc((size_t)F2 * KP2 * 2));
  __bf16* h2b   = (__bf16*)(ws + alloc((size_t)N_NODES * F2 * 2));
  float*  s2s   = (float*) (ws + alloc((size_t)N_NODES * 4));
  float*  s2d   = (float*) (ws + alloc((size_t)N_NODES * 4));
  float*  m2    = (float*) (ws + alloc((size_t)N_NODES * 4));
  float*  den2  = (float*) (ws + alloc((size_t)N_NODES * 4));
  float*  esc2  = (float*) (ws + alloc((size_t)N_ETOT * 4));
  float*  agg2  = (float*) (ws + alloc((size_t)N_NODES * F2 * 4));
  float*  gpool = (float*) (ws + alloc((size_t)NB * F2 * 4));
  float*  gout  = (float*) (ws + alloc((size_t)NB * F2 * 4));
  float*  xtb   = (float*) (ws + alloc((size_t)NB * PLMD * 4));
  float*  xcat  = (float*) (ws + alloc((size_t)NB * 256 * 4));
  float*  xc1   = (float*) (ws + alloc((size_t)NB * 1024 * 4));
  float*  xc2   = (float*) (ws + alloc((size_t)NB * 256 * 4));

  const unsigned T = 256;

  // ---- init (every call; ws may hold stale data)
  hipMemsetAsync(den1, 0, (size_t)N_NODES * NH1 * 4, stream);
  hipMemsetAsync(agg1, 0, (size_t)N_NODES * LD1 * 2, stream);
  hipMemsetAsync(den2, 0, (size_t)N_NODES * 4, stream);
  hipMemsetAsync(agg2, 0, (size_t)N_NODES * F2 * 4, stream);
  hipMemsetAsync(gpool, 0, (size_t)NB * F2 * 4, stream);
  fill_f32_k<<<gridFor((size_t)N_NODES * NH1, T), T, 0, stream>>>(m1, -1e30f, (size_t)N_NODES * NH1);
  fill_f32_k<<<gridFor((size_t)N_NODES, T), T, 0, stream>>>(m2, -1e30f, (size_t)N_NODES);

  // ---- pack operands to bf16 (B pre-transposed, K zero-padded)
  pack_x_k <<<gridFor((size_t)N_NODES * KP1, T), T, 0, stream>>>(x, XB);
  pack_w1_k<<<gridFor((size_t)LD1 * KP1, T), T, 0, stream>>>(W1, WT1);
  pack_w2_k<<<gridFor((size_t)F2 * KP2, T), T, 0, stream>>>(W2, WT2);

  // ---- GATConv1: h1 = x @ W1  (WMMA bf16, f32 accum; B slab staged in LDS)
  {
    int mtiles = N_NODES / 16, ntiles = LD1 / 16, ktiles = KP1 / 32;
    unsigned grid = (unsigned)(((mtiles + 7) / 8) * ntiles);
    wmma_gemm_bf16<<<grid, 256, 16 * KP1 * 2, stream>>>(
        XB, WT1, h1b, mtiles, ntiles, ktiles, KP1, KP1, LD1);
  }
  dots1_k  <<<gridFor((size_t)N_NODES * NH1, T), T, 0, stream>>>(h1b, a1s, a1d, ssrc1, sdst1);
  scores1_k<<<gridFor((size_t)N_ETOT * NH1, T), T, 0, stream>>>(ei, ssrc1, sdst1, esc1, m1);
  exp1_k   <<<gridFor((size_t)N_ETOT * NH1, T), T, 0, stream>>>(ei, esc1, m1, den1);
  agg1_k   <<<gridFor((size_t)N_ETOT * (HF1 / 2), T), T, 0, stream>>>(ei, esc1, den1, h1b, agg1);
  finalize1_k<<<gridFor((size_t)N_NODES * KP2, T), T, 0, stream>>>(agg1, b1, out1b);

  // ---- GATConv2: h2 = elu_out @ W2
  {
    int mtiles = N_NODES / 16, ntiles = F2 / 16, ktiles = KP2 / 32;
    unsigned grid = (unsigned)(((mtiles + 7) / 8) * ntiles);
    wmma_gemm_bf16<<<grid, 256, 16 * KP2 * 2, stream>>>(
        out1b, WT2, h2b, mtiles, ntiles, ktiles, KP2, KP2, F2);
  }
  dots2_k  <<<gridFor((size_t)N_NODES, T), T, 0, stream>>>(h2b, a2s, a2d, s2s, s2d);
  scores2_k<<<gridFor((size_t)N_ETOT, T), T, 0, stream>>>(ei, s2s, s2d, esc2, m2);
  exp2_k   <<<gridFor((size_t)N_ETOT, T), T, 0, stream>>>(ei, esc2, m2, den2);
  agg2_k   <<<gridFor((size_t)N_ETOT * F2, T), T, 0, stream>>>(ei, esc2, den2, h2b, agg2);
  pool_k   <<<gridFor((size_t)N_NODES * F2, T), T, 0, stream>>>(agg2, b2, batch, gpool);

  // ---- heads
  fc_relu_k<<<gridFor((size_t)NB * F2, T), T, 0, stream>>>(gpool, Wg, bg, gout, NB, F2, F2);
  plm_k    <<<gridFor((size_t)NB * PLMD, T), T, 0, stream>>>(te, Wxt, bxt, gamma, beta, rmean, rvar, xtb);
  concat_k <<<gridFor((size_t)NB * 256, T), T, 0, stream>>>(gout, xtb, xcat);
  fc_relu_k<<<gridFor((size_t)NB * 1024, T), T, 0, stream>>>(xcat, Wf1, bf1, xc1, NB, 256, 1024);
  fc_relu_k<<<gridFor((size_t)NB * 256, T), T, 0, stream>>>(xc1, Wf2, bf2, xc2, NB, 1024, 256);
  out_k    <<<gridFor((size_t)NB, T), T, 0, stream>>>(xc2, Wo, bo, out);
}
